// WaveCascadeTransformer_55989193671044
// MI455X (gfx1250) — 8-bit matrix operand order repaired and checked by reading, compiled, not run on hardware
//
#include <hip/hip_runtime.h>
#include <math.h>

#define BSZ   16
#define SEQ   2048
#define DIM   1024
#define VOUT  40
#define MROWS (BSZ * SEQ)
#define KBLK  (DIM / 64)
#define MTILES (MROWS / 16)
#define NTILES (DIM / 16)

typedef __attribute__((ext_vector_type(8))) int v8i;

__device__ __forceinline__ v8i load_tile_frag(const signed char* tile, int lane) {
  const int4* q = reinterpret_cast<const int4*>(tile);
  int4 a = q[lane];
  int4 b = q[32 + lane];
  v8i f;
  f[0] = a.x; f[1] = a.y; f[2] = a.z; f[3] = a.w;
  f[4] = b.x; f[5] = b.y; f[6] = b.z; f[7] = b.w;
  return f;
}

__device__ __forceinline__ float sigm(float x) { return 1.0f / (1.0f + expf(-x)); }
__device__ __forceinline__ float q_scale(float a) { return (a == 0.0f) ? 1.0f : a * (1.0f / 127.0f); }

__global__ void zero_f32(float* p, int n) {
  int i = blockIdx.x * blockDim.x + threadIdx.x;
  if (i < n) p[i] = 0.0f;
}

__global__ void pack_sign(const float* __restrict__ W, int* __restrict__ Q) {
  int tid  = blockIdx.x * blockDim.x + threadIdx.x;
  int tile = tid >> 8;
  int w    = tid & 255;
  int L, v;
  if (w < 128) { L = w >> 2; v = w & 3; }
  else         { int wl = w - 128; L = wl >> 2; v = 4 + (wl & 3); }
  int nb = tile >> 4, kb = tile & 15;
  int n  = nb * 16 + (L & 15);
  int hi = L >> 4;
  int kl = (v >> 1) * 16 + hi * 8 + (v & 1) * 4;
  float4 wv = *reinterpret_cast<const float4*>(W + (size_t)n * DIM + kb * 64 + kl);
  float xs[4] = {wv.x, wv.y, wv.z, wv.w};
  int packed = 0;
#pragma unroll
  for (int j = 0; j < 4; j++) {
    int s = (xs[j] > 0.0f) ? 1 : ((xs[j] < 0.0f) ? -1 : 0);
    packed |= (s & 0xFF) << (8 * j);
  }
  Q[tid] = packed;
}

__global__ __launch_bounds__(256) void rowstats(
    const float* __restrict__ X, float* __restrict__ mean, float* __restrict__ rstd,
    float* __restrict__ amax, int per_t) {
  __shared__ float sred[256];
  int row = blockIdx.x;
  int t   = threadIdx.x;
  float4 xv = reinterpret_cast<const float4*>(X + (size_t)row * DIM)[t];

  sred[t] = xv.x + xv.y + xv.z + xv.w;
  __syncthreads();
  for (int o = 128; o > 0; o >>= 1) { if (t < o) sred[t] += sred[t + o]; __syncthreads(); }
  float mu = sred[0] * (1.0f / DIM);
  __syncthreads();

  float d0 = xv.x - mu, d1 = xv.y - mu, d2 = xv.z - mu, d3 = xv.w - mu;
  sred[t] = d0 * d0 + d1 * d1 + d2 * d2 + d3 * d3;
  __syncthreads();
  for (int o = 128; o > 0; o >>= 1) { if (t < o) sred[t] += sred[t + o]; __syncthreads(); }
  float var = sred[0] * (1.0f / DIM);
  float r   = 1.0f / sqrtf(fmaxf(var + 1e-8f, 1e-10f));
  __syncthreads();

  float mx = fmaxf(fmaxf(fabsf(d0), fabsf(d1)), fmaxf(fabsf(d2), fabsf(d3))) * r;
  sred[t] = mx;
  __syncthreads();
  for (int o = 128; o > 0; o >>= 1) { if (t < o) sred[t] = fmaxf(sred[t], sred[t + o]); __syncthreads(); }

  if (t == 0) {
    mean[row] = mu;
    rstd[row] = r;
    int idx = per_t ? (row & (SEQ - 1)) : 0;
    atomicMax(reinterpret_cast<int*>(amax) + idx, __float_as_int(sred[0]));
  }
}

__global__ void pack_quant(
    const float* __restrict__ X, const float* __restrict__ mean,
    const float* __restrict__ rstd, int use_norm,
    const float* __restrict__ amax, int per_t, int* __restrict__ Q) {
  int tid  = blockIdx.x * blockDim.x + threadIdx.x;
  int tile = tid >> 8;
  int w    = tid & 255;
  int L, v;
  if (w < 128) { L = w >> 2; v = w & 3; }
  else         { int wl = w - 128; L = wl >> 2; v = 4 + (wl & 3); }
  int mb  = tile >> 4, kb = tile & 15;
  int row = mb * 16 + (L & 15);
  int hi  = L >> 4;
  int kbase = kb * 64 + (v >> 1) * 16 + hi * 8 + (v & 1) * 4;

  float mu = use_norm ? mean[row] : 0.0f;
  float r  = use_norm ? rstd[row] : 1.0f;
  float s  = q_scale(amax[per_t ? (row & (SEQ - 1)) : 0]);
  float4 xv = *reinterpret_cast<const float4*>(X + (size_t)row * DIM + kbase);
  float xs[4] = {xv.x, xv.y, xv.z, xv.w};
  int packed = 0;
#pragma unroll
  for (int j = 0; j < 4; j++) {
    float xn = (xs[j] - mu) * r;
    float qf = rintf(xn / s);
    qf = fminf(fmaxf(qf, -128.0f), 127.0f);
    packed |= (((int)qf) & 0xFF) << (8 * j);
  }
  Q[tid] = packed;
}

__global__ __launch_bounds__(256) void gemm_i8_tri(
    const signed char* __restrict__ Aq,
    const signed char* __restrict__ B0, const signed char* __restrict__ B1,
    const signed char* __restrict__ B2,
    const float* __restrict__ bias0, const float* __restrict__ bias1,
    const float* __restrict__ bias2, const float* __restrict__ amax,
    float* __restrict__ out0, float* __restrict__ out1, float* __restrict__ out2) {
  int wave = threadIdx.x >> 5, lane = threadIdx.x & 31;
  int mt   = blockIdx.x * 8 + wave;
  int nt0  = blockIdx.y * 4;

  v8i acc0[4], acc1[4], acc2[4];
#pragma unroll
  for (int i = 0; i < 4; i++) {
    acc0[i] = (v8i){0, 0, 0, 0, 0, 0, 0, 0};
    acc1[i] = (v8i){0, 0, 0, 0, 0, 0, 0, 0};
    acc2[i] = (v8i){0, 0, 0, 0, 0, 0, 0, 0};
  }

  const signed char* Abase = Aq + (size_t)mt * 16 * 1024;
  for (int kb = 0; kb < KBLK; kb++) {
    __builtin_prefetch(Abase + (kb + 1) * 1024, 0, 3);
    v8i a = load_tile_frag(Abase + kb * 1024, lane);
#pragma unroll
    for (int ti = 0; ti < 4; ti++) {
      size_t boff = ((size_t)(nt0 + ti) * 16 + kb) * 1024;
      v8i b0 = load_tile_frag(B0 + boff, lane);
      acc0[ti] = __builtin_amdgcn_wmma_i32_16x16x64_iu8(true, a, true, b0, acc0[ti], false, false);
      v8i b1 = load_tile_frag(B1 + boff, lane);
      acc1[ti] = __builtin_amdgcn_wmma_i32_16x16x64_iu8(true, a, true, b1, acc1[ti], false, false);
      v8i b2 = load_tile_frag(B2 + boff, lane);
      acc2[ti] = __builtin_amdgcn_wmma_i32_16x16x64_iu8(true, a, true, b2, acc2[ti], false, false);
    }
  }

  int hi = lane >> 4, nl = lane & 15;
  float scl[8];
#pragma unroll
  for (int i = 0; i < 8; i++)
    scl[i] = q_scale(amax[(mt * 16 + i + 8 * hi) & (SEQ - 1)]);

#pragma unroll
  for (int ti = 0; ti < 4; ti++) {
    int col = (nt0 + ti) * 16 + nl;
    float bv0 = bias0[col], bv1 = bias1[col], bv2 = bias2[col];
#pragma unroll
    for (int i = 0; i < 8; i++) {
      size_t rowg = (size_t)(mt * 16 + i + 8 * hi);
      __builtin_nontemporal_store((float)acc0[ti][i] * scl[i] + bv0, out0 + rowg * DIM + col);
      __builtin_nontemporal_store((float)acc1[ti][i] * scl[i] + bv1, out1 + rowg * DIM + col);
      __builtin_nontemporal_store((float)acc2[ti][i] * scl[i] + bv2, out2 + rowg * DIM + col);
    }
  }
}

__global__ __launch_bounds__(256) void gemm_i8(
    const signed char* __restrict__ Aq, const signed char* __restrict__ Bq,
    const float* __restrict__ bias, const float* __restrict__ amax,
    float* __restrict__ out) {
  int wave = threadIdx.x >> 5, lane = threadIdx.x & 31;
  int mt   = blockIdx.x * 8 + wave;
  int nt0  = blockIdx.y * 4;

  v8i acc[4];
#pragma unroll
  for (int i = 0; i < 4; i++) acc[i] = (v8i){0, 0, 0, 0, 0, 0, 0, 0};

  const signed char* Abase = Aq + (size_t)mt * 16 * 1024;
  for (int kb = 0; kb < KBLK; kb++) {
    __builtin_prefetch(Abase + (kb + 1) * 1024, 0, 3);
    v8i a = load_tile_frag(Abase + kb * 1024, lane);
#pragma unroll
    for (int ti = 0; ti < 4; ti++) {
      v8i b = load_tile_frag(Bq + ((size_t)(nt0 + ti) * 16 + kb) * 1024, lane);
      acc[ti] = __builtin_amdgcn_wmma_i32_16x16x64_iu8(true, a, true, b, acc[ti], false, false);
    }
  }

  int hi = lane >> 4, nl = lane & 15;
  float s = q_scale(amax[0]);
#pragma unroll
  for (int ti = 0; ti < 4; ti++) {
    int col = (nt0 + ti) * 16 + nl;
    float bv = bias[col];
#pragma unroll
    for (int i = 0; i < 8; i++) {
      size_t rowg = (size_t)(mt * 16 + i + 8 * hi);
      out[rowg * DIM + col] = (float)acc[ti][i] * s + bv;
    }
  }
}

__global__ __launch_bounds__(256) void gemm_i8_dual_glu(
    const signed char* __restrict__ Aq, const signed char* __restrict__ Bg,
    const signed char* __restrict__ Bu, const float* __restrict__ bg,
    const float* __restrict__ bu, const float* __restrict__ amaxIn,
    float* __restrict__ out, float* __restrict__ amaxOut) {
  __shared__ float red[256];
  int wave = threadIdx.x >> 5, lane = threadIdx.x & 31;
  int mt   = blockIdx.x * 8 + wave;
  int nt0  = blockIdx.y * 4;

  v8i accg[4], accu[4];
#pragma unroll
  for (int i = 0; i < 4; i++) {
    accg[i] = (v8i){0, 0, 0, 0, 0, 0, 0, 0};
    accu[i] = (v8i){0, 0, 0, 0, 0, 0, 0, 0};
  }

  const signed char* Abase = Aq + (size_t)mt * 16 * 1024;
  for (int kb = 0; kb < KBLK; kb++) {
    __builtin_prefetch(Abase + (kb + 1) * 1024, 0, 3);
    v8i a = load_tile_frag(Abase + kb * 1024, lane);
#pragma unroll
    for (int ti = 0; ti < 4; ti++) {
      size_t boff = ((size_t)(nt0 + ti) * 16 + kb) * 1024;
      v8i b0 = load_tile_frag(Bg + boff, lane);
      accg[ti] = __builtin_amdgcn_wmma_i32_16x16x64_iu8(true, a, true, b0, accg[ti], false, false);
      v8i b1 = load_tile_frag(Bu + boff, lane);
      accu[ti] = __builtin_amdgcn_wmma_i32_16x16x64_iu8(true, a, true, b1, accu[ti], false, false);
    }
  }

  int hi = lane >> 4, nl = lane & 15;
  float s = q_scale(amaxIn[0]);
  float lmax = 0.0f;
#pragma unroll
  for (int ti = 0; ti < 4; ti++) {
    int col = (nt0 + ti) * 16 + nl;
    float bgv = bg[col], buv = bu[col];
#pragma unroll
    for (int i = 0; i < 8; i++) {
      size_t rowg = (size_t)(mt * 16 + i + 8 * hi);
      float g  = (float)accg[ti][i] * s + bgv;
      float u  = (float)accu[ti][i] * s + buv;
      float gu = (g * sigm(g)) * u;
      out[rowg * DIM + col] = gu;
      lmax = fmaxf(lmax, fabsf(gu));
    }
  }
  red[threadIdx.x] = lmax;
  __syncthreads();
  for (int o = 128; o > 0; o >>= 1) {
    if (threadIdx.x < o) red[threadIdx.x] = fmaxf(red[threadIdx.x], red[threadIdx.x + o]);
    __syncthreads();
  }
  if (threadIdx.x == 0)
    atomicMax(reinterpret_cast<int*>(amaxOut), __float_as_int(red[0]));
}

__global__ __launch_bounds__(256) void scan_k(
    float* __restrict__ F, const float* __restrict__ C, const float* __restrict__ G) {
  int tid = blockIdx.x * blockDim.x + threadIdx.x;
  int b   = tid >> 10;
  int k   = tid & 1023;
  size_t base = (size_t)b * SEQ * DIM + k;
  float h = 0.0f;
#pragma unroll 2
  for (int t = 0; t < SEQ; t++) {
    size_t idx = base + (size_t)t * DIM;
    float f = __builtin_nontemporal_load(F + idx);
    float c = __builtin_nontemporal_load(C + idx);
    float g = __builtin_nontemporal_load(G + idx);
    float sf = sigm(f);
    h = sf * h + (1.0f - sf) * (c * sigm(c));
    F[idx] = h * sigm(g);
  }
}

__global__ __launch_bounds__(256) void out_proj(
    const float* __restrict__ Dm, const float* __restrict__ mean,
    const float* __restrict__ rstd, const float* __restrict__ amax,
    const float* __restrict__ Wout, const float* __restrict__ bout,
    float* __restrict__ out) {
  __shared__ float ybuf[DIM];
  __shared__ float red[256];
  int row = blockIdx.x;
  int t   = threadIdx.x;
  float s  = q_scale(amax[0]);
  float mu = mean[row], r = rstd[row];
  float4 dv = reinterpret_cast<const float4*>(Dm + (size_t)row * DIM)[t];
  float xs[4] = {dv.x, dv.y, dv.z, dv.w};
  float4 yv;
  float* yp = &yv.x;
#pragma unroll
  for (int j = 0; j < 4; j++) {
    float xn = (xs[j] - mu) * r;
    float qf = fminf(fmaxf(rintf(xn / s), -128.0f), 127.0f);
    yp[j] = qf * s;
  }
  reinterpret_cast<float4*>(ybuf)[t] = yv;
  __syncthreads();

  int v = t & 63, half = t >> 6;
  float partial = 0.0f;
  if (v < VOUT) {
    const float* wr = Wout + (size_t)v * DIM + half * 256;
    const float* yb = ybuf + half * 256;
#pragma unroll 4
    for (int h = 0; h < 256; h++) partial += yb[h] * wr[h];
  }
  red[t] = partial;
  __syncthreads();
  if (t < 64 && v < VOUT)
    out[(size_t)row * VOUT + v] = red[t] + red[t + 64] + red[t + 128] + red[t + 192] + bout[v];
}

extern "C" void kernel_launch(void* const* d_in, const int* in_sizes, int n_in,
                              void* d_out, int out_size, void* d_ws, size_t ws_size,
                              hipStream_t stream) {
  (void)in_sizes; (void)n_in; (void)out_size; (void)ws_size;
  const float* x    = (const float*)d_in[0];
  const float* Wmats[6] = {(const float*)d_in[1], (const float*)d_in[2], (const float*)d_in[3],
                           (const float*)d_in[7], (const float*)d_in[8], (const float*)d_in[9]};
  const float* b_f  = (const float*)d_in[4];
  const float* b_c  = (const float*)d_in[5];
  const float* b_g  = (const float*)d_in[6];
  const float* bg2  = (const float*)d_in[10];
  const float* bu2  = (const float*)d_in[11];
  const float* bd2  = (const float*)d_in[12];
  const float* Wout = (const float*)d_in[13];
  const float* bout = (const float*)d_in[14];
  float* out = (float*)d_out;

  char* ws = (char*)d_ws;
  size_t off = 0;
  auto alloc = [&](size_t bytes) -> char* {
    char* p = ws + off;
    off += (bytes + 255) & ~(size_t)255;
    return p;
  };
  signed char* sw[6];
  for (int i = 0; i < 6; i++) sw[i] = (signed char*)alloc((size_t)DIM * DIM);
  float* amax  = (float*)alloc((SEQ + 8) * sizeof(float));
  float* meanb = (float*)alloc((size_t)MROWS * 4);
  float* rstdb = (float*)alloc((size_t)MROWS * 4);
  signed char* XQ = (signed char*)alloc((size_t)MROWS * DIM);
  float* BUFA = (float*)alloc((size_t)MROWS * DIM * 4);
  float* BUFB = (float*)alloc((size_t)MROWS * DIM * 4);
  float* BUFC = (float*)alloc((size_t)MROWS * DIM * 4);

  zero_f32<<<(SEQ + 8 + 255) / 256, 256, 0, stream>>>(amax, SEQ + 8);

  const int sdw = DIM * DIM / 4;
  for (int i = 0; i < 6; i++)
    pack_sign<<<sdw / 256, 256, 0, stream>>>(Wmats[i], (int*)sw[i]);

  rowstats<<<MROWS, 256, 0, stream>>>(x, meanb, rstdb, amax, 1);
  const int adw = MROWS * DIM / 4;
  pack_quant<<<adw / 256, 256, 0, stream>>>(x, meanb, rstdb, 1, amax, 1, (int*)XQ);

  dim3 ggrid(MTILES / 8, NTILES / 4);
  gemm_i8_tri<<<ggrid, 256, 0, stream>>>(XQ, sw[0], sw[1], sw[2],
                                         b_f, b_c, b_g, amax, BUFA, BUFB, BUFC);

  scan_k<<<(BSZ * DIM) / 256, 256, 0, stream>>>(BUFA, BUFB, BUFC);

  rowstats<<<MROWS, 256, 0, stream>>>(BUFA, meanb, rstdb, amax + SEQ, 0);
  pack_quant<<<adw / 256, 256, 0, stream>>>(BUFA, meanb, rstdb, 1, amax + SEQ, 0, (int*)XQ);
  gemm_i8_dual_glu<<<ggrid, 256, 0, stream>>>(XQ, sw[3], sw[4], bg2, bu2,
                                              amax + SEQ, BUFB, amax + SEQ + 1);
  pack_quant<<<adw / 256, 256, 0, stream>>>(BUFB, meanb, rstdb, 0, amax + SEQ + 1, 0, (int*)XQ);
  gemm_i8<<<ggrid, 256, 0, stream>>>(XQ, sw[5], bd2, amax + SEQ + 1, BUFC);

  rowstats<<<MROWS, 256, 0, stream>>>(BUFC, meanb, rstdb, amax + SEQ + 2, 0);
  out_proj<<<MROWS, 256, 0, stream>>>(BUFC, meanb, rstdb, amax + SEQ + 2, Wout, bout, out);
}
